// AdaptiveSparsityAttention_37864431681669
// MI455X (gfx1250) — matrix operand order repaired and checked by reading, not compiled, not run on hardware
//
#include <hip/hip_runtime.h>
#include <hip/hip_bf16.h>

// ---------------------------------------------------------------------------
// AdaptiveSparsityAttention for MI455X (gfx1250), wave32 + WMMA f16->f32,
// with TDM (tensor_load_to_lds) weight-slab prefetch in the GEMMs.
// B=4, S=2048, H=2048, NH=16, HD=128.
// ---------------------------------------------------------------------------

#define BB   4
#define SS   2048
#define HH   2048
#define NH   16
#define HD   128
#define MM   (BB * SS)            // 8192 rows
#define TIDX 409                  // int(S*0.2) clamped

typedef __attribute__((ext_vector_type(16))) _Float16 v16h;
typedef __attribute__((ext_vector_type(8)))  _Float16 v8h;
typedef __attribute__((ext_vector_type(8)))  float    v8f;
typedef __attribute__((ext_vector_type(4)))  unsigned int u32x4;
typedef __attribute__((ext_vector_type(8)))  int      i32x8;
typedef __attribute__((ext_vector_type(4)))  int      i32x4;

__device__ __forceinline__ v8h ld8(const _Float16* p) { return *(const v8h*)p; }

__device__ __forceinline__ v16h cat8(v8h lo, v8h hi) {
  v16h r;
#pragma unroll
  for (int i = 0; i < 8; ++i) { r[i] = lo[i]; r[i + 8] = hi[i]; }
  return r;
}

__device__ __forceinline__ v8f wmma_f16(v16h a, v16h b, v8f c) {
  // D = A x B + C, 16x16x32 f16 inputs, f32 accumulate.
  return __builtin_amdgcn_wmma_f32_16x16x32_f16(
      /*neg_a=*/false, a, /*neg_b=*/false, b,
      /*c_mod=*/(short)0, c, /*reuse_a=*/false, /*reuse_b=*/false);
}

// ---------------------------------------------------------------------------
// TDM prefetch: pull a 128(N-rows) x 32(K, contiguous halves) slab of the
// transposed weight matrix (row stride 2048 halves) into an LDS landing pad.
// Descriptor per ISA 8.3-8.5: count=1; type=2; data_size=2B; tensor 2048x2048;
// tile_dim0=32, tile_dim1=128; tensor_dim0_stride=2048. Pure prefetch: the
// landing buffer is never read, so numerics are independent of the DMA.
// ---------------------------------------------------------------------------
__device__ __forceinline__ void tdm_prefetch_slab(const _Float16* gaddr,
                                                  unsigned lds_off) {
  unsigned long long ga = (unsigned long long)(uintptr_t)gaddr;
  u32x4 g0;
  g0[0] = 1u;                                                 // count=1
  g0[1] = lds_off;                                            // lds_addr
  g0[2] = (unsigned)(ga & 0xffffffffu);                       // global_addr lo
  g0[3] = (unsigned)((ga >> 32) & 0x01ffffffu) | (2u << 30);  // addr hi | type=2
  i32x8 g1;
  g1[0] = 1 << 16;                        // wg_mask=0, data_size=2B
  g1[1] = (int)((2048u & 0xffffu) << 16); // tensor_dim0[15:0] in bits[63:48]
  g1[2] = (int)(2048u << 16);             // dim0[31:16]=0 | tensor_dim1[15:0]
  g1[3] = (int)(32u << 16);               // dim1[31:16]=0 | tile_dim0=32
  g1[4] = 128;                            // tile_dim1=128, tile_dim2=0
  g1[5] = 2048;                           // tensor_dim0_stride[31:0]
  g1[6] = 0;                              // stride hi | dim1_stride lo
  g1[7] = 0;
  i32x4 z4 = {0, 0, 0, 0};
#if defined(__clang_major__) && __clang_major__ >= 23
  i32x8 z8 = {0, 0, 0, 0, 0, 0, 0, 0};
  __builtin_amdgcn_tensor_load_to_lds(g0, g1, z4, z4, z8, 0);
#else
  __builtin_amdgcn_tensor_load_to_lds(g0, g1, z4, z4, 0);
#endif
}

// ---------------------------------------------------------------------------
// Kernel 1: convert f32 weight (K x N row-major) -> f16 transposed (N x K).
// ---------------------------------------------------------------------------
__global__ __launch_bounds__(256) void convert_wt(const float* __restrict__ W,
                                                  _Float16* __restrict__ Wt) {
  size_t i = (size_t)blockIdx.x * 256 + threadIdx.x;   // over H*H
  int k = (int)(i >> 11);        // row in W
  int n = (int)(i & (HH - 1));   // col in W
  Wt[(size_t)n * HH + k] = (_Float16)W[i];
}

// ---------------------------------------------------------------------------
// Kernel 2: importance GEMV fused with x f32->f16 conversion.
// ---------------------------------------------------------------------------
__global__ __launch_bounds__(256) void importance_convert(
    const float* __restrict__ x, const float* __restrict__ Wi,
    const float* __restrict__ bi, _Float16* __restrict__ xh,
    float* __restrict__ imp) {
  __shared__ float red[256];
  const int r = blockIdx.x;
  const int t = threadIdx.x;
  const float* xr = x + (size_t)r * HH;
  _Float16* xo = xh + (size_t)r * HH;
  float s = 0.f;
#pragma unroll
  for (int i = 0; i < 8; ++i) {
    int c = t + i * 256;
    float v = xr[c];
    xo[c] = (_Float16)v;
    s += v * Wi[c];
  }
  red[t] = s;
  __syncthreads();
  for (int o = 128; o > 0; o >>= 1) {
    if (t < o) red[t] += red[t + o];
    __syncthreads();
  }
  if (t == 0) imp[r] = red[0] + bi[0];
}

// ---------------------------------------------------------------------------
// Kernel 3: per-batch bitonic sort of 2048 importances -> threshold -> mask.
// sorted_desc[409] == sorted_asc[1638].
// ---------------------------------------------------------------------------
__global__ __launch_bounds__(1024) void threshold_mask(
    const float* __restrict__ imp, float* __restrict__ mask) {
  __shared__ float arr[SS];
  __shared__ float thr;
  const int b = blockIdx.x;
  const int t = threadIdx.x;
  const float* ib = imp + (size_t)b * SS;
  arr[t] = ib[t];
  arr[t + 1024] = ib[t + 1024];
  __syncthreads();
  for (unsigned k = 2; k <= SS; k <<= 1) {
    for (unsigned j = k >> 1; j > 0; j >>= 1) {
#pragma unroll
      for (int p = 0; p < 2; ++p) {
        unsigned i = (unsigned)t + (unsigned)p * 1024u;
        unsigned ixj = i ^ j;
        if (ixj > i) {
          bool up = ((i & k) == 0);       // ascending segment
          float a = arr[i], c = arr[ixj];
          if ((a > c) == up) { arr[i] = c; arr[ixj] = a; }
        }
      }
      __syncthreads();
    }
  }
  if (t == 0) thr = arr[SS - 1 - TIDX];
  __syncthreads();
  const float th = thr;
  mask[(size_t)b * SS + t] = (ib[t] > th) ? 1.f : 0.f;
  mask[(size_t)b * SS + t + 1024] = (ib[t + 1024] > th) ? 1.f : 0.f;
}

// ---------------------------------------------------------------------------
// Kernel 4: WMMA GEMM  C[M x N] = A[M x K(f16)] * Bt[N x K(f16)] + bias[n],
// optionally * mask[row].  Macro-tile 128x128, 8 waves (4x2), each wave
// 32x64 = 2x4 tiles.  Wave 0 TDM-prefetches the next weight K-slab.
// K-loop force-unrolled 2x: 16 WMMAs per iteration for load/WMMA pipelining.
// MODE 0: f16 out; MODE 1: f16 out transposed per batch (V); MODE 2: f32 out.
// ---------------------------------------------------------------------------
template <int MODE>
__global__ __launch_bounds__(256) void gemm_wmma(
    const _Float16* __restrict__ A, const _Float16* __restrict__ Bt,
    const float* __restrict__ bias, const float* __restrict__ mask,
    _Float16* __restrict__ outh, float* __restrict__ outf) {
  __shared__ _Float16 tdm_buf[128 * 32];   // TDM landing pad (prefetch only)
  const int lane = threadIdx.x & 31;
  const int w = threadIdx.x >> 5;
  const int m16 = lane & 15;
  const int hi8 = (lane >> 4) * 8;
  const int hi16 = (lane >> 4) * 16;
  const int bm = blockIdx.y * 128 + (w & 3) * 32;   // wave M origin
  const int bn = blockIdx.x * 128 + (w >> 2) * 64;  // wave N origin
  const unsigned tdm_off = (unsigned)(uintptr_t)(void*)tdm_buf;
  const bool tdm_wave = (w == 0);
  const _Float16* wslab = Bt + (size_t)(blockIdx.x * 128) * HH;

  v8f acc[2][4];
#pragma unroll
  for (int i = 0; i < 2; ++i)
#pragma unroll
    for (int j = 0; j < 4; ++j) acc[i][j] = {};

  const _Float16* ap0 = A + (size_t)(bm + m16) * HH;
  const _Float16* ap1 = A + (size_t)(bm + 16 + m16) * HH;
  const _Float16* bp[4];
#pragma unroll
  for (int nt = 0; nt < 4; ++nt)
    bp[nt] = Bt + (size_t)(bn + nt * 16 + m16) * HH;

#pragma unroll 2
  for (int k0 = 0; k0 < HH; k0 += 32) {
    if (tdm_wave && (k0 & 63) == 0 && k0 + 64 < HH) {
      // TDM: stream a future 128x32 weight slab toward LDS/L2 while we compute.
      tdm_prefetch_slab(wslab + (k0 + 64), tdm_off);
    }
    __builtin_prefetch((const void*)(ap0 + k0 + 512), 0, 3);
    v16h a0 = cat8(ld8(ap0 + k0 + hi8), ld8(ap0 + k0 + 16 + hi8));
    v16h a1 = cat8(ld8(ap1 + k0 + hi8), ld8(ap1 + k0 + 16 + hi8));
#pragma unroll
    for (int nt = 0; nt < 4; ++nt) {
      v16h b = cat8(ld8(bp[nt] + k0 + hi8), ld8(bp[nt] + k0 + 16 + hi8));
      acc[0][nt] = wmma_f16(a0, b, acc[0][nt]);
      acc[1][nt] = wmma_f16(a1, b, acc[1][nt]);
    }
  }
  if (tdm_wave) __builtin_amdgcn_s_wait_tensorcnt(0);  // drain TDM before exit

#pragma unroll
  for (int mt = 0; mt < 2; ++mt) {
#pragma unroll
    for (int nt = 0; nt < 4; ++nt) {
      const int col = bn + nt * 16 + m16;
      const float bv = bias[col];
      const int rowbase = bm + mt * 16 + hi8;
#pragma unroll
      for (int r = 0; r < 8; ++r) {
        const int row = rowbase + r;
        float v = acc[mt][nt][r] + bv;
        if (mask) v *= mask[row];
        if (MODE == 2) {
          outf[(size_t)row * HH + col] = v;
        } else if (MODE == 1) {
          const int bb = row >> 11, s = row & (SS - 1);
          outh[((size_t)(bb * HH + col)) * SS + s] = (_Float16)v;
        } else {
          outh[(size_t)row * HH + col] = (_Float16)v;
        }
      }
    }
  }
}

// ---------------------------------------------------------------------------
// Kernel 5: flash attention, dense over zeroed K/V (matches reference).
// Block = 8 waves x 16 queries = 128 queries per (b, h).
// ---------------------------------------------------------------------------
__global__ __launch_bounds__(256) void attn_wmma(
    const _Float16* __restrict__ qh, const _Float16* __restrict__ kh,
    const _Float16* __restrict__ vt, _Float16* __restrict__ oh) {
  __shared__ _Float16 pbuf[8][16 * 32];   // wave-private 16x32 P tile
  const int lane = threadIdx.x & 31;
  const int w = threadIdx.x >> 5;
  const int m16 = lane & 15;
  const int hi8 = (lane >> 4) * 8;
  const int hi16 = (lane >> 4) * 16;
  const int b = blockIdx.z, h = blockIdx.y;
  const int q0 = blockIdx.x * 128 + w * 16;

  const _Float16* qp = qh + ((size_t)(b * SS + q0 + m16)) * HH + h * HD;
  const _Float16* kp = kh + ((size_t)b * SS) * HH + h * HD;
  const _Float16* vp = vt + ((size_t)(b * HH + h * HD)) * SS;

  v16h qf[4];
#pragma unroll
  for (int c = 0; c < 4; ++c)
    qf[c] = cat8(ld8(qp + c * 32 + hi8), ld8(qp + c * 32 + 16 + hi8));

  v8f acc[8];
#pragma unroll
  for (int t = 0; t < 8; ++t) acc[t] = {};
  float mrun[8], lrun[8], corr[8];
#pragma unroll
  for (int r = 0; r < 8; ++r) { mrun[r] = -1e30f; lrun[r] = 0.f; }

  const float scale = 0.08838834764831845f;  // 1/sqrt(128)

  for (int key0 = 0; key0 < SS; key0 += 32) {
    v8f s0 = {}, s1 = {};
    const _Float16* kb0 = kp + (size_t)(key0 + m16) * HH;
    const _Float16* kb1 = kp + (size_t)(key0 + 16 + m16) * HH;
#pragma unroll
    for (int c = 0; c < 4; ++c) {
      v16h b0 = cat8(ld8(kb0 + c * 32 + hi8), ld8(kb0 + c * 32 + 16 + hi8));
      s0 = wmma_f16(qf[c], b0, s0);
      v16h b1 = cat8(ld8(kb1 + c * 32 + hi8), ld8(kb1 + c * 32 + 16 + hi8));
      s1 = wmma_f16(qf[c], b1, s1);
    }
    // online softmax per row (rows split across 16-lane halves)
#pragma unroll
    for (int r = 0; r < 8; ++r) {
      float x0 = s0[r] * scale, x1 = s1[r] * scale;
      float mx = fmaxf(x0, x1);
#pragma unroll
      for (int d = 1; d < 16; d <<= 1) mx = fmaxf(mx, __shfl_xor(mx, d, 32));
      float mnew = fmaxf(mrun[r], mx);
      float cr = __expf(mrun[r] - mnew);
      float p0 = __expf(x0 - mnew);
      float p1 = __expf(x1 - mnew);
      float ps = p0 + p1;
#pragma unroll
      for (int d = 1; d < 16; d <<= 1) ps += __shfl_xor(ps, d, 32);
      lrun[r] = lrun[r] * cr + ps;
      mrun[r] = mnew;
      corr[r] = cr;
      const int m = r + hi8;
      pbuf[w][m * 32 + m16] = (_Float16)p0;        // keys 0..15
      pbuf[w][m * 32 + 16 + m16] = (_Float16)p1;   // keys 16..31
    }
#pragma unroll
    for (int t = 0; t < 8; ++t)
#pragma unroll
      for (int r = 0; r < 8; ++r) acc[t][r] *= corr[r];

    // Re-fragment P as WMMA A operand (wave-private LDS, in-order DS).
    v16h pf = cat8(ld8(&pbuf[w][m16 * 32 + hi8]),
                   ld8(&pbuf[w][m16 * 32 + 16 + hi8]));
#pragma unroll
    for (int t = 0; t < 8; ++t) {
      const _Float16* vb = vp + (size_t)(t * 16 + m16) * SS + key0;
      v16h bv = cat8(ld8(vb + hi8), ld8(vb + 16 + hi8));
      acc[t] = wmma_f16(pf, bv, acc[t]);
    }
  }

  _Float16* op = oh + ((size_t)(b * SS + q0)) * HH + h * HD;
#pragma unroll
  for (int t = 0; t < 8; ++t)
#pragma unroll
    for (int r = 0; r < 8; ++r) {
      float v = acc[t][r] / lrun[r];
      op[(size_t)(r + hi8) * HH + t * 16 + m16] = (_Float16)v;
    }
}

// ---------------------------------------------------------------------------
// Host launch
// ---------------------------------------------------------------------------
extern "C" void kernel_launch(void* const* d_in, const int* in_sizes, int n_in,
                              void* d_out, int out_size, void* d_ws,
                              size_t ws_size, hipStream_t stream) {
  const float* x  = (const float*)d_in[0];
  const float* Wq = (const float*)d_in[1];
  const float* bq = (const float*)d_in[2];
  const float* Wk = (const float*)d_in[3];
  const float* bk = (const float*)d_in[4];
  const float* Wv = (const float*)d_in[5];
  const float* bv = (const float*)d_in[6];
  const float* Wo = (const float*)d_in[7];
  const float* bo = (const float*)d_in[8];
  const float* Wi = (const float*)d_in[9];
  const float* bi = (const float*)d_in[10];

  char* ws = (char*)d_ws;
  const size_t XH = (size_t)MM * HH * sizeof(_Float16);       // 32 MB
  const size_t WH = (size_t)HH * HH * sizeof(_Float16);       // 8 MB
  _Float16* x_h  = (_Float16*)(ws);
  _Float16* wq_h = (_Float16*)(ws + XH);
  _Float16* wk_h = (_Float16*)(ws + XH + WH);
  _Float16* wv_h = (_Float16*)(ws + XH + 2 * WH);
  _Float16* wo_h = (_Float16*)(ws + XH + 3 * WH);
  _Float16* q_h  = (_Float16*)(ws + XH + 4 * WH);
  _Float16* k_h  = (_Float16*)(ws + 2 * XH + 4 * WH);
  _Float16* v_t  = (_Float16*)(ws + 3 * XH + 4 * WH);
  float* imp     = (float*)(ws + 4 * XH + 4 * WH);
  float* mask    = (float*)(ws + 4 * XH + 4 * WH + (size_t)MM * 4);
  _Float16* attn_h = x_h;  // alias: x_h dead after the V GEMM

  const int wgrid = (HH * HH) / 256;
  convert_wt<<<wgrid, 256, 0, stream>>>(Wq, wq_h);
  convert_wt<<<wgrid, 256, 0, stream>>>(Wk, wk_h);
  convert_wt<<<wgrid, 256, 0, stream>>>(Wv, wv_h);
  convert_wt<<<wgrid, 256, 0, stream>>>(Wo, wo_h);

  importance_convert<<<MM, 256, 0, stream>>>(x, Wi, bi, x_h, imp);
  threshold_mask<<<BB, 1024, 0, stream>>>(imp, mask);

  dim3 gg(HH / 128, MM / 128);
  gemm_wmma<0><<<gg, 256, 0, stream>>>(x_h, wq_h, bq, nullptr, q_h, nullptr);
  gemm_wmma<0><<<gg, 256, 0, stream>>>(x_h, wk_h, bk, mask, k_h, nullptr);
  gemm_wmma<1><<<gg, 256, 0, stream>>>(x_h, wv_h, bv, mask, v_t, nullptr);

  attn_wmma<<<dim3(SS / 128, NH, BB), 256, 0, stream>>>(q_h, k_h, v_t, attn_h);

  gemm_wmma<2><<<gg, 256, 0, stream>>>(attn_h, wo_h, bo, nullptr, nullptr,
                                       (float*)d_out);
}
